// RecurentNeuralNetwork_1958505087099
// MI455X (gfx1250) — matrix operand order repaired and checked by reading, not compiled, not run on hardware
//
#include <hip/hip_runtime.h>
#include <hip/hip_bf16.h>

typedef __attribute__((ext_vector_type(16))) _Float16 v16h;
typedef __attribute__((ext_vector_type(8)))  _Float16 v8h;
typedef __attribute__((ext_vector_type(8)))  float    v8f;

#define HDIM   32
#define GDIM   128          // 4*H
#define SEQ    512
#define IN0    19
#define NBATCH 4096
#define TILE_M 16
#define NWAVE  4            // waves per block
#define ODIM   30

// ---- fast nonlinearities: prefer the CDNA5 v_tanh_f32 trans op ----
#if __has_builtin(__builtin_amdgcn_tanhf)
__device__ __forceinline__ float tanh_f(float x) { return __builtin_amdgcn_tanhf(x); }
#elif __has_builtin(__builtin_amdgcn_tanh_f32)
__device__ __forceinline__ float tanh_f(float x) { return __builtin_amdgcn_tanh_f32(x); }
#else
__device__ __forceinline__ float tanh_f(float x) {
    return 2.0f * __builtin_amdgcn_rcpf(1.0f + __expf(-2.0f * x)) - 1.0f;
}
#endif
// sigmoid(x) = 0.5*tanh(x/2) + 0.5  (1 trans op + mul + fma)
__device__ __forceinline__ float sigm_f(float x) {
    return __builtin_fmaf(0.5f, tanh_f(0.5f * x), 0.5f);
}

__device__ __forceinline__ v8f wmma_f16(v16h a, v16h b, v8f c) {
    return __builtin_amdgcn_wmma_f32_16x16x32_f16(false, a, false, b,
                                                  (short)0, c, false, false);
}

// Fused 2-layer LSTM + FC.  One wave processes a 16-batch tile across all 512
// steps; every matmul is a v_wmma_f32_16x16x32_f16 (K == H == 32).
// Per step: 2 layers x 2 hidden-halves x 4 gate tiles x 2 chained WMMAs = 32 WMMA.
__global__ __launch_bounds__(32 * NWAVE)
void lstm2_fused_kernel(const float* __restrict__ x,
                        const float* __restrict__ Wih0, const float* __restrict__ Whh0,
                        const float* __restrict__ bih0, const float* __restrict__ bhh0,
                        const float* __restrict__ Wih1, const float* __restrict__ Whh1,
                        const float* __restrict__ bih1, const float* __restrict__ bhh1,
                        const float* __restrict__ fcw,  const float* __restrict__ fcb,
                        float* __restrict__ out)
{
    // 32 weight tiles in WMMA B layout (lane n = column, K = 8*hi + p for p<8, 16 + 8*hi + (p-8) for p>=8):
    //   tiles 0..7   Wih0^T (K padded 19->32 with zeros)
    //   tiles 8..15  Whh0^T
    //   tiles 16..23 Wih1^T
    //   tiles 24..31 Whh1^T
    __shared__ __align__(32) _Float16 s_w[32 * 512];
    __shared__ float s_b0[GDIM];
    __shared__ float s_b1[GDIM];
    // per-wave D-layout -> A-layout transpose staging for h0 / h1 (row-major 16x32 f16)
    __shared__ __align__(32) _Float16 s_h[NWAVE][2][TILE_M * HDIM];

    const int tid = threadIdx.x;

    if (tid < GDIM) {
        s_b0[tid] = bih0[tid] + bhh0[tid];
        s_b1[tid] = bih1[tid] + bhh1[tid];
    }

    for (int e = tid; e < 32 * 512; e += blockDim.x) {
        const int tile = e >> 9;
        const int r    = e & 511;
        const int ln   = r >> 4;        // lane this element belongs to
        const int p    = r & 15;        // element within lane (VGPR half index)
        const int n    = ln & 15;       // B column
        const int k    = (p < 8) ? (((ln >> 4) << 3) + p) : (16 + ((ln >> 4) << 3) + (p - 8));   // K = 8*hi + p (p<8), 16 + 8*hi + (p-8) (p>=8)
        const int g    = tile & 7;      // gate tile (16 gate columns)
        float v;
        if (tile < 8)       v = (k < IN0) ? Wih0[(g * 16 + n) * IN0 + k] : 0.0f;
        else if (tile < 16) v = Whh0[(g * 16 + n) * HDIM + k];
        else if (tile < 24) v = Wih1[(g * 16 + n) * HDIM + k];
        else                v = Whh1[(g * 16 + n) * HDIM + k];
        s_w[e] = (_Float16)v;
    }
    __syncthreads();

    const int wave = tid >> 5;
    const int lane = tid & 31;
    const int m    = lane & 15;         // batch row within tile / D column
    const int hi   = lane >> 4;
    const int batch_base = (blockIdx.x * NWAVE + wave) * TILE_M;

    // bias per gate tile: all 8 accumulator VGPRs of a lane hold column m
    float b0v[8], b1v[8];
#pragma unroll
    for (int g = 0; g < 8; ++g) {
        b0v[g] = s_b0[g * 16 + m];
        b1v[g] = s_b1[g * 16 + m];
    }

    _Float16* h0s = &s_h[wave][0][0];
    _Float16* h1s = &s_h[wave][1][0];

    v16h h0A = {};                      // h state in WMMA A layout (f16)
    v16h h1A = {};
    float c0[16] = {};                  // cell state, D-tile layout per lane
    float c1[16] = {};

    const float* xrow = x + (size_t)(batch_base + m) * SEQ * IN0;

    for (int t = 0; t < SEQ; ++t) {
        // ---- load x tile (16 x 19, zero-padded to K=32) in A layout ----
        v16h xa;
#pragma unroll
        for (int p = 0; p < 16; ++p) {
            const int k = (p < 8) ? (hi * 8 + p) : (16 + hi * 8 + (p - 8));
            xa[p] = (k < IN0) ? (_Float16)xrow[k] : (_Float16)0.0f;
        }
        __builtin_prefetch(xrow + IN0, 0, 3);   // next timestep, same row

        // ---- layer 0: gates = x*Wih0^T + h0*Whh0^T + bias ----
        // Two half passes: hidden half hh uses gate tiles {hh,2+hh,4+hh,6+hh};
        // only 4 accumulator tiles live at a time (keeps VGPRs < 256).
#pragma unroll
        for (int hh = 0; hh < 2; ++hh) {
            v8f acc[4];
#pragma unroll
            for (int q = 0; q < 4; ++q) {
                const int g = 2 * q + hh;
                const float b = b0v[g];
                v8f c = {b, b, b, b, b, b, b, b};
                const v16h wih = *(const v16h*)(s_w + g * 512 + lane * 16);
                c = wmma_f16(xa, wih, c);
                const v16h whh = *(const v16h*)(s_w + (8 + g) * 512 + lane * 16);
                c = wmma_f16(h0A, whh, c);
                acc[q] = c;
            }
#pragma unroll
            for (int j = 0; j < 8; ++j) {
                const int e = hh * 8 + j;
                const float ig = sigm_f(acc[0][j]);
                const float fg = sigm_f(acc[1][j]);
                const float gg = tanh_f(acc[2][j]);
                const float og = sigm_f(acc[3][j]);
                const float c  = fg * c0[e] + ig * gg;
                c0[e] = c;
                const float h = og * tanh_f(c);
                h0s[(j + hi * 8) * HDIM + hh * 16 + m] = (_Float16)h;
            }
        }
        // transpose read-back: D layout -> A layout (wave-private LDS, in-order DS)
        {
            const v8h lo = *(const v8h*)(h0s + m * HDIM + hi * 8);
            const v8h hh8 = *(const v8h*)(h0s + m * HDIM + 16 + hi * 8);
#pragma unroll
            for (int p = 0; p < 8; ++p) { h0A[p] = lo[p]; h0A[8 + p] = hh8[p]; }
        }

        // ---- layer 1: gates = h0*Wih1^T + h1*Whh1^T + bias ----
#pragma unroll
        for (int hh = 0; hh < 2; ++hh) {
            v8f acc[4];
#pragma unroll
            for (int q = 0; q < 4; ++q) {
                const int g = 2 * q + hh;
                const float b = b1v[g];
                v8f c = {b, b, b, b, b, b, b, b};
                const v16h wih = *(const v16h*)(s_w + (16 + g) * 512 + lane * 16);
                c = wmma_f16(h0A, wih, c);
                const v16h whh = *(const v16h*)(s_w + (24 + g) * 512 + lane * 16);
                c = wmma_f16(h1A, whh, c);
                acc[q] = c;
            }
#pragma unroll
            for (int j = 0; j < 8; ++j) {
                const int e = hh * 8 + j;
                const float ig = sigm_f(acc[0][j]);
                const float fg = sigm_f(acc[1][j]);
                const float gg = tanh_f(acc[2][j]);
                const float og = sigm_f(acc[3][j]);
                const float c  = fg * c1[e] + ig * gg;
                c1[e] = c;
                const float h = og * tanh_f(c);
                h1s[(j + hi * 8) * HDIM + hh * 16 + m] = (_Float16)h;
            }
        }
        {
            const v8h lo = *(const v8h*)(h1s + m * HDIM + hi * 8);
            const v8h hh8 = *(const v8h*)(h1s + m * HDIM + 16 + hi * 8);
#pragma unroll
            for (int p = 0; p < 8; ++p) { h1A[p] = lo[p]; h1A[8 + p] = hh8[p]; }
        }

        xrow += IN0;
    }

    // ---- final FC: out(16x30) = h1_last(16x32) @ fcw^T + fcb ----
    // h1_last sits row-major in h1s.  480 outputs / 32 lanes = 15 each.
#pragma unroll
    for (int q = 0; q < 15; ++q) {
        const int idx = lane * 15 + q;       // 0..479
        const int b   = idx / ODIM;
        const int o   = idx % ODIM;
        float s = fcb[o];
#pragma unroll
        for (int k = 0; k < HDIM; ++k)
            s += (float)h1s[b * HDIM + k] * fcw[o * HDIM + k];
        out[(size_t)(batch_base + b) * ODIM + o] = s;
    }
}

extern "C" void kernel_launch(void* const* d_in, const int* in_sizes, int n_in,
                              void* d_out, int out_size, void* d_ws, size_t ws_size,
                              hipStream_t stream) {
    (void)in_sizes; (void)n_in; (void)out_size; (void)d_ws; (void)ws_size;
    const float* x    = (const float*)d_in[0];
    const float* Wih0 = (const float*)d_in[1];
    const float* Whh0 = (const float*)d_in[2];
    const float* bih0 = (const float*)d_in[3];
    const float* bhh0 = (const float*)d_in[4];
    const float* Wih1 = (const float*)d_in[5];
    const float* Whh1 = (const float*)d_in[6];
    const float* bih1 = (const float*)d_in[7];
    const float* bhh1 = (const float*)d_in[8];
    const float* fcw  = (const float*)d_in[9];
    const float* fcb  = (const float*)d_in[10];
    float* out = (float*)d_out;

    dim3 grid(NBATCH / (TILE_M * NWAVE));   // 64 blocks
    dim3 block(32 * NWAVE);                 // 128 threads = 4 waves
    lstm2_fused_kernel<<<grid, block, 0, stream>>>(
        x, Wih0, Whh0, bih0, bhh0, Wih1, Whh1, bih1, bhh1, fcw, fcb, out);
}
